// LSTM_18116172054677
// MI455X (gfx1250) — matrix operand order repaired and checked by reading, not compiled, not run on hardware
//
#include <hip/hip_runtime.h>
#include <hip/hip_bf16.h>
#include <math.h>

// ---------------------------------------------------------------------------
// LSTM rollout (T=128, B=256, I=H=D=1024) + per-timestep BatchNorm.
// bf16 WMMA (v_wmma_f32_16x16x32_bf16) for all GEMMs, fp32 state/accum.
// Weights converted once to bf16 (18MB) -> resident in 192MB L2.
// Epilogue uses hardware v_tanh_f32 (CDNA5 TRANS op) for all activations.
// K-loops unrolled x2 so alternating fragment register sets hide the
// WMMA->VALU WAR hazard NOPs behind independent loads.
// ---------------------------------------------------------------------------

typedef __bf16 bf16_t;
typedef __attribute__((ext_vector_type(8)))  __bf16 v8bf;
typedef __attribute__((ext_vector_type(16))) __bf16 v16bf;
typedef __attribute__((ext_vector_type(8)))  float  v8f;

#define TSTEPS 128
#define BATCH  256
#define HDIM   1024

#if __has_builtin(__builtin_amdgcn_tanhf)
__device__ __forceinline__ float fast_tanh(float x) { return __builtin_amdgcn_tanhf(x); }
#else
__device__ __forceinline__ float fast_tanh(float x) {
    // branch-free fallback: 1 - 2/(e^{2x}+1); exp never negative -> no NaN path
    return 1.0f - 2.0f * __builtin_amdgcn_rcpf(__expf(2.0f * x) + 1.0f);
}
#endif
__device__ __forceinline__ float sigf(float x) { return 0.5f + 0.5f * fast_tanh(0.5f * x); }

// A fragment: 16x32 bf16 tile, rows m0..m0+15, cols kk..kk+31 of row-major src (ld=HDIM).
// Layout: lane&15 = M; lanes<16 hold K 0-15;
// lanes>=16 hold K 16-31 (each as 2x8 contiguous bf16).
__device__ __forceinline__ v16bf load_a_frag(const bf16_t* __restrict__ src,
                                             int m0, int kk, int lane) {
    const bf16_t* p = src + (size_t)(m0 + (lane & 15)) * HDIM + kk + ((lane >> 4) << 4);
    v8bf lo = *(const v8bf*)p;
    v8bf hi = *(const v8bf*)(p + 8);
    v16bf a;
#pragma unroll
    for (int i = 0; i < 8; ++i) { a[i] = lo[i]; a[i + 8] = hi[i]; }
    return a;
}

// B fragment: 32x16 tile of W^T, i.e. weight rows (=N) j0..j0+15, cols kk..kk+31
// of row-major W (ld=HDIM). Layout: lane&15 = N; lanes<16 hold K 0-15,
// lanes>=16 hold K 16-31 (16 contiguous bf16 = one 32B load).
__device__ __forceinline__ v16bf load_b_frag(const bf16_t* __restrict__ W,
                                             int j0, int kk, int lane) {
    const bf16_t* p = W + (size_t)(j0 + (lane & 15)) * HDIM + kk + ((lane >> 4) << 4);
    return *(const v16bf*)p;
}

// ---------------------------------------------------------------------------
// gates = x @ W_ih^T + h @ W_hh^T + biases ; fused LSTM cell update.
// Block = 4 waves (128 thr). Wave tile: 32(M)x16(N), all 4 gates
// (8 accumulators -> every B fragment reused twice, A reused across gates).
// Block tile: BM=32, BN=64. Grid: (256/32, 1024/64) = (8, 16).
// ---------------------------------------------------------------------------
__global__ __launch_bounds__(128) void lstm_gates_kernel(
    const bf16_t* __restrict__ xb, const bf16_t* __restrict__ hb,
    const bf16_t* __restrict__ Wih, const bf16_t* __restrict__ Whh,
    const float* __restrict__ b_ih, const float* __restrict__ b_hh,
    float* __restrict__ cbuf, bf16_t* __restrict__ hout)
{
    const int w    = threadIdx.x >> 5;     // 0..3
    const int lane = threadIdx.x & 31;
    const int m0 = blockIdx.x * 32;        // rows m0..m0+31 (two 16-row tiles)
    const int n0 = blockIdx.y * 64 + w * 16;

    v8f acc[4][2];
#pragma unroll
    for (int g = 0; g < 4; ++g)
#pragma unroll
        for (int mt = 0; mt < 2; ++mt)
#pragma unroll
            for (int e = 0; e < 8; ++e) acc[g][mt][e] = 0.0f;

#pragma unroll 1
    for (int phase = 0; phase < 2; ++phase) {
        const bf16_t* src = phase ? hb  : xb;
        const bf16_t* W   = phase ? Whh : Wih;
#pragma unroll 2
        for (int kk = 0; kk < HDIM; kk += 32) {
            v16bf a0 = load_a_frag(src, m0,      kk, lane);
            v16bf a1 = load_a_frag(src, m0 + 16, kk, lane);
#pragma unroll
            for (int g = 0; g < 4; ++g) {
                v16bf b = load_b_frag(W, g * HDIM + n0, kk, lane);
                acc[g][0] = __builtin_amdgcn_wmma_f32_16x16x32_bf16(
                    false, a0, false, b, (short)0, acc[g][0], false, false);
                acc[g][1] = __builtin_amdgcn_wmma_f32_16x16x32_bf16(
                    false, a1, false, b, (short)0, acc[g][1], false, false);
            }
        }
    }

    // Fused bias + LSTM elementwise (all TRANS-op activations, no branches).
    // C/D layout: elem r -> M = r + 8*(lane>=16), N = lane&15.
    const int col   = n0 + (lane & 15);
    const int rhalf = (lane >> 4) << 3;
    const float bi0 = b_ih[0 * HDIM + col] + b_hh[0 * HDIM + col];
    const float bi1 = b_ih[1 * HDIM + col] + b_hh[1 * HDIM + col];
    const float bi2 = b_ih[2 * HDIM + col] + b_hh[2 * HDIM + col];
    const float bi3 = b_ih[3 * HDIM + col] + b_hh[3 * HDIM + col];
#pragma unroll
    for (int mt = 0; mt < 2; ++mt) {
#pragma unroll
        for (int r = 0; r < 8; ++r) {
            const int row = m0 + mt * 16 + rhalf + r;
            const size_t idx = (size_t)row * HDIM + col;
            const float gi = acc[0][mt][r] + bi0;
            const float gf = acc[1][mt][r] + bi1;
            const float gg = acc[2][mt][r] + bi2;
            const float go = acc[3][mt][r] + bi3;
            const float cn = sigf(gf) * cbuf[idx] + sigf(gi) * fast_tanh(gg);
            const float hn = sigf(go) * fast_tanh(cn);
            cbuf[idx] = cn;
            hout[idx] = (bf16_t)hn;
        }
    }
}

// ---------------------------------------------------------------------------
// y = h @ W_lin^T + b_lin ; writes fp32 y to d_out[t] and bf16 y as next x.
// Block = 4 waves. Wave tile 32x16 (2 accumulators, B reused twice).
// Grid (8, 16).
// ---------------------------------------------------------------------------
__global__ __launch_bounds__(128) void lstm_out_kernel(
    const bf16_t* __restrict__ hb, const bf16_t* __restrict__ Wlin,
    const float* __restrict__ b_lin,
    float* __restrict__ yout, bf16_t* __restrict__ xb)
{
    const int w    = threadIdx.x >> 5;
    const int lane = threadIdx.x & 31;
    const int m0 = blockIdx.x * 32;
    const int n0 = blockIdx.y * 64 + w * 16;

    v8f acc[2];
#pragma unroll
    for (int mt = 0; mt < 2; ++mt)
#pragma unroll
        for (int e = 0; e < 8; ++e) acc[mt][e] = 0.0f;

#pragma unroll 2
    for (int kk = 0; kk < HDIM; kk += 32) {
        v16bf a0 = load_a_frag(hb, m0,      kk, lane);
        v16bf a1 = load_a_frag(hb, m0 + 16, kk, lane);
        v16bf b  = load_b_frag(Wlin, n0, kk, lane);
        acc[0] = __builtin_amdgcn_wmma_f32_16x16x32_bf16(
            false, a0, false, b, (short)0, acc[0], false, false);
        acc[1] = __builtin_amdgcn_wmma_f32_16x16x32_bf16(
            false, a1, false, b, (short)0, acc[1], false, false);
    }

    const int col   = n0 + (lane & 15);
    const int rhalf = (lane >> 4) << 3;
    const float bias = b_lin[col];
#pragma unroll
    for (int mt = 0; mt < 2; ++mt) {
#pragma unroll
        for (int r = 0; r < 8; ++r) {
            const int row = m0 + mt * 16 + rhalf + r;
            const float y = acc[mt][r] + bias;
            yout[(size_t)row * HDIM + col] = y;
            xb[(size_t)row * HDIM + col]   = (bf16_t)y;
        }
    }
}

// ---------------------------------------------------------------------------
// BatchNorm1d (affine=False, training): normalize over batch per (t, d),
// in place on d_out. One wave32 per (t, d) column; 8 columns per block.
// ---------------------------------------------------------------------------
__global__ __launch_bounds__(256) void bn_kernel(float* __restrict__ ys)
{
    const int idx  = blockIdx.x * 8 + (threadIdx.x >> 5);
    const int lane = threadIdx.x & 31;
    const int t = idx >> 10;
    const int d = idx & 1023;
    float* base = ys + (size_t)t * BATCH * HDIM + d;

    float v[8];
    float s = 0.0f, s2 = 0.0f;
#pragma unroll
    for (int j = 0; j < 8; ++j) {
        v[j] = base[(size_t)(lane + 32 * j) * HDIM];
        s  += v[j];
        s2 += v[j] * v[j];
    }
#pragma unroll
    for (int m = 16; m >= 1; m >>= 1) {
        s  += __shfl_xor(s,  m, 32);
        s2 += __shfl_xor(s2, m, 32);
    }
    const float mean = s * (1.0f / BATCH);
    const float var  = s2 * (1.0f / BATCH) - mean * mean;
    const float inv  = rsqrtf(var + 1e-5f);
#pragma unroll
    for (int j = 0; j < 8; ++j)
        base[(size_t)(lane + 32 * j) * HDIM] = (v[j] - mean) * inv;
}

__global__ void cvt_f32_to_bf16(const float* __restrict__ in, bf16_t* __restrict__ out, int n)
{
    const int i = blockIdx.x * blockDim.x + threadIdx.x;
    if (i < n) out[i] = (bf16_t)in[i];
}

// ---------------------------------------------------------------------------
// Workspace layout (bytes):
//   Wih_b  @ 0         : 8388608
//   Whh_b  @ 8388608   : 8388608
//   Wlin_b @ 16777216  : 2097152
//   xb     @ 18874368  : 524288
//   hb0    @ 19398656  : 524288
//   hb1    @ 19922944  : 524288
//   cbuf   @ 20447232  : 1048576   (total ~20.5 MB)
// ---------------------------------------------------------------------------
extern "C" void kernel_launch(void* const* d_in, const int* in_sizes, int n_in,
                              void* d_out, int out_size, void* d_ws, size_t ws_size,
                              hipStream_t stream)
{
    (void)in_sizes; (void)n_in; (void)out_size; (void)ws_size;

    const float* inputs = (const float*)d_in[0];
    const float* W_ih   = (const float*)d_in[1];
    const float* W_hh   = (const float*)d_in[2];
    const float* b_ih   = (const float*)d_in[3];
    const float* b_hh   = (const float*)d_in[4];
    const float* W_lin  = (const float*)d_in[5];
    const float* b_lin  = (const float*)d_in[6];
    const float* h0     = (const float*)d_in[7];
    const float* c0     = (const float*)d_in[8];

    char* ws = (char*)d_ws;
    bf16_t* Wih_b  = (bf16_t*)(ws);
    bf16_t* Whh_b  = (bf16_t*)(ws + 8388608);
    bf16_t* Wlin_b = (bf16_t*)(ws + 16777216);
    bf16_t* xb     = (bf16_t*)(ws + 18874368);
    bf16_t* hb0    = (bf16_t*)(ws + 19398656);
    bf16_t* hb1    = (bf16_t*)(ws + 19922944);
    float*  cbuf   = (float*) (ws + 20447232);

    const int NW  = 4 * HDIM * HDIM;   // 4M elems (W_ih, W_hh)
    const int NL  = HDIM * HDIM;       // 1M elems (W_lin)
    const int NBH = BATCH * HDIM;      // 256K elems

    cvt_f32_to_bf16<<<(NW  + 255) / 256, 256, 0, stream>>>(W_ih,   Wih_b,  NW);
    cvt_f32_to_bf16<<<(NW  + 255) / 256, 256, 0, stream>>>(W_hh,   Whh_b,  NW);
    cvt_f32_to_bf16<<<(NL  + 255) / 256, 256, 0, stream>>>(W_lin,  Wlin_b, NL);
    cvt_f32_to_bf16<<<(NBH + 255) / 256, 256, 0, stream>>>(inputs, xb,     NBH);
    cvt_f32_to_bf16<<<(NBH + 255) / 256, 256, 0, stream>>>(h0,     hb0,    NBH);
    hipMemcpyAsync(cbuf, c0, (size_t)NBH * sizeof(float), hipMemcpyDeviceToDevice, stream);

    float* out = (float*)d_out;
    for (int t = 0; t < TSTEPS; ++t) {
        const bf16_t* hr = (t & 1) ? hb1 : hb0;  // h read (double-buffered)
        bf16_t*       hw = (t & 1) ? hb0 : hb1;  // h write
        lstm_gates_kernel<<<dim3(8, 16), 128, 0, stream>>>(
            xb, hr, Wih_b, Whh_b, b_ih, b_hh, cbuf, hw);
        lstm_out_kernel<<<dim3(8, 16), 128, 0, stream>>>(
            hw, Wlin_b, b_lin, out + (size_t)t * BATCH * HDIM, xb);
    }

    // 128*1024 columns, 8 per block
    bn_kernel<<<(TSTEPS * HDIM) / 8, 256, 0, stream>>>(out);
}
